// CuriosityModule_63024350101868
// MI455X (gfx1250) — compile-verified, matrix operand order checked by reading, not run on hardware
//
#include <hip/hip_runtime.h>
#include <math.h>

// Problem constants (from reference setup_inputs)
#define B_ROWS 4096
#define DD 64
#define NH 8
#define HH 128
#define CC 16384
#define TT 2
#define TAU_F 0.3f
#define XS 136   // padded LDS row stride (ushorts) to spread banks

typedef __attribute__((ext_vector_type(16))) __bf16 v16bf;
typedef __attribute__((ext_vector_type(8)))  float  v8f;

union BFrag { v16bf v; unsigned short u[16]; };
union AccU  { v8f   v; float          f[8];  };

__device__ __forceinline__ unsigned short f32_to_bf16(float x) {
  unsigned int u = __float_as_uint(x);
  unsigned int r = (u + 0x7FFFu + ((u >> 16) & 1u)) >> 16;  // RNE
  return (unsigned short)r;
}

// K index for element e of a 16-element bf16 fragment (ISA 7.12.2, 16-bit A 16x32):
// lanes 0-15 hold K in {0..7,16..23}; lanes 16-31 hold K in {8..15,24..31};
// each VGPR packs two consecutive K. f selects a 32-wide K slice.
__device__ __forceinline__ int frag_k(int f, int e, int half) {
  int j  = e >> 1;
  int kb = (j < 4) ? (2 * j) : (16 + 2 * (j - 4));
  return f * 32 + kb + half * 8 + (e & 1);
}

// Build a B-fragment (K-slice f) for this lane from an LDS row-major bf16
// matrix with row stride XS. Lane's column n = nl.
__device__ __forceinline__ v16bf build_bfrag(const unsigned short* sbase,
                                             int nl, int half, int f) {
  BFrag r;
#pragma unroll
  for (int e = 0; e < 16; ++e) r.u[e] = sbase[nl * XS + frag_k(f, e, half)];
  return r.v;
}

// ---- CDNA5 async global->LDS copy (ASYNCcnt path), per ISA 15.18.3 ----
__device__ __forceinline__ void async_ld_b128(unsigned lds_off, unsigned long long gaddr) {
  asm volatile("global_load_async_to_lds_b128 %0, %1, off"
               :: "v"(lds_off), "v"(gaddr) : "memory");
}
__device__ __forceinline__ void wait_async_le(int) {
  asm volatile("s_wait_asynccnt 0x4" ::: "memory");
}
__device__ __forceinline__ void wait_async_zero() {
  asm volatile("s_wait_asynccnt 0x0" ::: "memory");
}

// ---------------- K0a: P = sum_h Wk_h^T @ mem_h  (64x64) ----------------
__global__ void k0a_P(const float* __restrict__ Wk, const float* __restrict__ mem,
                      float* __restrict__ P) {
  int j = blockIdx.x, e = threadIdx.x;
  float acc = 0.f;
  for (int h = 0; h < NH; ++h)
    for (int dd = 0; dd < DD; ++dd)
      acc += Wk[(h * DD + dd) * DD + j] * mem[(size_t)(h * DD + dd) * DD + e];
  P[j * DD + e] = acc;
}

// ---------------- K0b: M = (1/H) * P @ Wo^T  (64x64) ----------------
__global__ void k0b_M(const float* __restrict__ P, const float* __restrict__ Wo,
                      float* __restrict__ M) {
  int j = blockIdx.x, o = threadIdx.x;
  float acc = 0.f;
  for (int e = 0; e < DD; ++e) acc += P[j * DD + e] * Wo[o * DD + e];
  M[j * DD + o] = acc * (1.0f / NH);
}

// ---- K1: query = (z*hot)@Wq^T + bq ; qn(bf16) ; v_mem = query@M ----
__global__ void k1_query(const float* __restrict__ z, const float* __restrict__ hot,
                         const float* __restrict__ Wq, const float* __restrict__ bq,
                         const float* __restrict__ M,
                         unsigned short* __restrict__ qn_bf, float* __restrict__ v_mem) {
  __shared__ float s_zh[4][DD];
  __shared__ float s_q[4][DD];
  __shared__ float s_red[4][DD];
  int row = threadIdx.x >> 6;
  int c   = threadIdx.x & 63;
  int b   = blockIdx.x * 4 + row;
  s_zh[row][c] = z[(size_t)b * DD + c] * hot[(size_t)b * DD + c];
  __syncthreads();
  float q = bq[c];
  for (int j = 0; j < DD; ++j) q += s_zh[row][j] * Wq[c * DD + j];
  s_q[row][c]   = q;
  s_red[row][c] = q * q;
  __syncthreads();
  for (int s = 32; s > 0; s >>= 1) {
    if (c < s) s_red[row][c] += s_red[row][c + s];
    __syncthreads();
  }
  float scale = 1.0f / fmaxf(sqrtf(s_red[row][0]), 1e-8f);
  qn_bf[(size_t)b * DD + c] = f32_to_bf16(q * scale);
  float vm = 0.f;
  for (int j = 0; j < DD; ++j) vm += s_q[row][j] * M[j * DD + c];
  v_mem[(size_t)b * DD + c] = vm;
}

// ---- K2: kn = normalize(cache_keys) -> bf16 in WMMA A-fragment order ----
__global__ void k2_keys(const float* __restrict__ keys, unsigned short* __restrict__ knfrag) {
  __shared__ float s_sq[16][16];
  __shared__ float s_scale[16];
  int tile = blockIdx.x;
  int kl   = threadIdx.x >> 4;
  int part = threadIdx.x & 15;
  const float* krow = keys + (size_t)(tile * 16 + kl) * DD;
  float ss = 0.f;
  for (int d0 = 0; d0 < 4; ++d0) { float x = krow[part * 4 + d0]; ss += x * x; }
  s_sq[kl][part] = ss;
  __syncthreads();
  if (part == 0) {
    float t = 0.f;
    for (int p = 0; p < 16; ++p) t += s_sq[kl][p];
    s_scale[kl] = 1.0f / fmaxf(sqrtf(t), 1e-8f);
  }
  __syncthreads();
  for (int out = threadIdx.x; out < 1024; out += 256) {
    int lane = out >> 5, rem = out & 31;
    int f = rem >> 4, e = rem & 15;
    int half = lane >> 4, keyl = lane & 15;
    int K = frag_k(f, e, half);
    float val = keys[(size_t)(tile * 16 + keyl) * DD + K] * s_scale[keyl];
    knfrag[(size_t)tile * 1024 + out] = f32_to_bf16(val);
  }
}

// ---- K2w: generic weight (Rx128) -> bf16 A-fragment order; grid = R/16 tiles ----
__global__ void k2w_frag(const float* __restrict__ W, unsigned short* __restrict__ out) {
  int u = blockIdx.x;
  for (int o = threadIdx.x; o < 2048; o += 128) {
    int f = o >> 9, rem = o & 511, lane = rem >> 4, e = rem & 15;
    int half = lane >> 4, m = lane & 15;
    int K = frag_k(f & 1, e, half) + (f >> 1) * 64;  // f*32 via frag_k(f&1)+64*(f>>1)
    out[(size_t)u * 2048 + o] = f32_to_bf16(W[(size_t)(u * 16 + m) * HH + K]);
  }
}

// ---- K3: fused sims (WMMA bf16) + top-4 + episodic gather + v_combined ----
// Key tiles staged through LDS with double-buffered async global->LDS DMA.
__global__ void k3_sims(const unsigned short* __restrict__ qn_bf,
                        const unsigned short* __restrict__ knfrag,
                        const float* __restrict__ cache_vals,
                        const float* __restrict__ v_mem,
                        float* __restrict__ v_comb) {
  __shared__ unsigned short s_kt[2][4][1024];  // [buf][wave][2KB tile]
  __shared__ float lv[4][32][4];
  __shared__ int   li[4][32][4];
  __shared__ int   fi[16][4];
  int tid  = threadIdx.x;
  int w    = tid >> 5, lane = tid & 31;
  int half = lane >> 4, nl = lane & 15;
  int qrow = blockIdx.x * 16 + nl;

  BFrag b0, b1;
#pragma unroll
  for (int e = 0; e < 16; ++e) {
    b0.u[e] = qn_bf[(size_t)qrow * DD + frag_k(0, e, half)];
    b1.u[e] = qn_bf[(size_t)qrow * DD + frag_k(1, e, half)];
  }

  float tv[4] = {-1e30f, -1e30f, -1e30f, -1e30f};
  int   ti[4] = {0, 0, 0, 0};
  const v8f zero = {0.f, 0.f, 0.f, 0.f, 0.f, 0.f, 0.f, 0.f};

  // LDS byte addresses of this wave's two tile buffers (flat low 32 bits).
  unsigned lbase0 = (unsigned)(size_t)&s_kt[0][w][0];
  unsigned lbase1 = (unsigned)(size_t)&s_kt[1][w][0];

  int t0 = w * (CC / 16 / 4), t1 = t0 + (CC / 16 / 4);

  // preload tile t0 into buffer 0 (4 x 512B per wave)
  {
    unsigned long long g = (unsigned long long)(knfrag + (size_t)t0 * 1024);
#pragma unroll
    for (int i = 0; i < 4; ++i)
      async_ld_b128(lbase0 + i * 512 + lane * 16, g + i * 512 + lane * 16);
  }

  int p = 0;
  for (int t = t0; t < t1; ++t) {
    // issue next tile into the alternate buffer (wrapped so always valid)
    int tn = (t + 1 < t1) ? (t + 1) : t0;
    unsigned nb = p ? lbase0 : lbase1;
    unsigned long long g = (unsigned long long)(knfrag + (size_t)tn * 1024);
#pragma unroll
    for (int i = 0; i < 4; ++i)
      async_ld_b128(nb + i * 512 + lane * 16, g + i * 512 + lane * 16);

    // async loads complete in order: <=4 outstanding => current tile resident
    wait_async_le(4);

    const v16bf* lt = (const v16bf*)&s_kt[p][w][0];
    v16bf a0 = lt[lane * 2 + 0];   // K = 0..31
    v16bf a1 = lt[lane * 2 + 1];   // K = 32..63
    AccU acc; acc.v = zero;
    acc.v = __builtin_amdgcn_wmma_f32_16x16x32_bf16(false, a0, false, b0.v,
                                                    (short)0, acc.v, false, false);
    acc.v = __builtin_amdgcn_wmma_f32_16x16x32_bf16(false, a1, false, b1.v,
                                                    (short)0, acc.v, false, false);
    float maxv = fmaxf(fmaxf(fmaxf(acc.f[0], acc.f[1]), fmaxf(acc.f[2], acc.f[3])),
                       fmaxf(fmaxf(acc.f[4], acc.f[5]), fmaxf(acc.f[6], acc.f[7])));
    if (maxv > tv[3]) {                         // rare path
      int kbase = t * 16 + half * 8;
#pragma unroll
      for (int r = 0; r < 8; ++r) {
        float v = acc.f[r];
        int idx = kbase + r;
        if (v > tv[3]) {
          if (v > tv[0]) { tv[3]=tv[2];ti[3]=ti[2]; tv[2]=tv[1];ti[2]=ti[1];
                           tv[1]=tv[0];ti[1]=ti[0]; tv[0]=v;ti[0]=idx; }
          else if (v > tv[1]) { tv[3]=tv[2];ti[3]=ti[2]; tv[2]=tv[1];ti[2]=ti[1];
                                tv[1]=v;ti[1]=idx; }
          else if (v > tv[2]) { tv[3]=tv[2];ti[3]=ti[2]; tv[2]=v;ti[2]=idx; }
          else { tv[3]=v; ti[3]=idx; }
        }
      }
    }
    p ^= 1;
  }
  wait_async_zero();  // drain DMA before LDS-wide merge phase

  for (int s = 0; s < 4; ++s) { lv[w][lane][s] = tv[s]; li[w][lane][s] = ti[s]; }
  __syncthreads();

  if (tid < 16) {
    int r = tid;
    unsigned int used = 0;
    for (int s = 0; s < 4; ++s) {
      float best = -2e30f; int bi = 0, bslot = 0;
      for (int c = 0; c < 32; ++c) {
        if (used & (1u << c)) continue;
        int cw = c >> 3, rem = c & 7;
        int cl = r + ((rem >> 2) << 4);
        int cs = rem & 3;
        float v = lv[cw][cl][cs];
        if (v > best) { best = v; bi = li[cw][cl][cs]; bslot = c; }
      }
      used |= 1u << bslot;
      fi[r][s] = bi;
    }
  }
  __syncthreads();

  for (int out = tid; out < 16 * DD; out += 128) {
    int r = out >> 6, cdim = out & 63;
    int b = blockIdx.x * 16 + r;
    float s = 0.f;
    for (int k = 0; k < 4; ++k) s += cache_vals[(size_t)fi[r][k] * DD + cdim];
    v_comb[(size_t)b * DD + cdim] =
        0.5f * (v_mem[(size_t)b * DD + cdim] + 0.25f * s);
  }
}

// one gate slice: acc = bias + sum_f Wih_tile(f) x bx(f) + Whh_tile(f) x bh(f)
__device__ __forceinline__ void gate_slice(const v16bf* Aih, const v16bf* Ahh,
                                           const v16bf* bx, const v16bf* bh,
                                           int tile, int lane,
                                           const float* bias_i, const float* bias_h,
                                           int unit0, int half, float out[8]) {
  AccU ai, ah;
#pragma unroll
  for (int r = 0; r < 8; ++r) {
    int unit = unit0 + r + 8 * half;
    ai.f[r] = bias_i[unit];
    ah.f[r] = bias_h[unit];
  }
#pragma unroll
  for (int f = 0; f < 4; ++f) {
    v16bf wa = Aih[((size_t)tile * 4 + f) * 32 + lane];
    v16bf wh = Ahh[((size_t)tile * 4 + f) * 32 + lane];
    ai.v = __builtin_amdgcn_wmma_f32_16x16x32_bf16(false, wa, false, bx[f],
                                                   (short)0, ai.v, false, false);
    ah.v = __builtin_amdgcn_wmma_f32_16x16x32_bf16(false, wh, false, bh[f],
                                                   (short)0, ah.v, false, false);
  }
#pragma unroll
  for (int r = 0; r < 8; ++r) out[r] = ai.f[r] + ah.f[r];
}

// ---- K4: tensorized GRU rollout (bf16 WMMA, f32 accum), cf partials, f32 fusion ----
// Block: 16 batch rows (N of every WMMA), 128 threads = 4 waves.
// Wave w owns hidden-unit tiles {2w, 2w+1} (of 8) and znew tile w (of 4).
__global__ void k4_gru(const float* __restrict__ z, const float* __restrict__ gap_norm,
                       const float* __restrict__ v_comb, const float* __restrict__ act_emb,
                       const int* __restrict__ actions,
                       const unsigned short* __restrict__ wf_ih,
                       const unsigned short* __restrict__ wf_hh,
                       const unsigned short* __restrict__ wf_o1,
                       const unsigned short* __restrict__ wf_o2,
                       const float* __restrict__ b_ih, const float* __restrict__ b_hh,
                       const float* __restrict__ h0,
                       const float* __restrict__ b_out1, const float* __restrict__ b_out2,
                       const float* __restrict__ Wf, const float* __restrict__ bfv,
                       float* __restrict__ z_out, float* __restrict__ cf_part) {
  __shared__ unsigned short s_x[16 * XS];   // x = [z_prev | a] bf16
  __shared__ unsigned short s_h[16 * XS];   // h bf16
  __shared__ unsigned short s_g[16 * XS];   // gelu output bf16
  __shared__ float s_z0[16 * DD];
  __shared__ float s_vc[16 * DD];
  __shared__ float s_bih[3 * HH], s_bhh[3 * HH], s_bo1[HH], s_bo2[DD];
  __shared__ float s_red[128];

  int tid = threadIdx.x;
  int w = tid >> 5, lane = tid & 31, half = lane >> 4, nl = lane & 15;
  int blk = blockIdx.x;

  for (int i = tid; i < 3 * HH; i += 128) { s_bih[i] = b_ih[i]; s_bhh[i] = b_hh[i]; }
  if (tid < HH) s_bo1[tid] = b_out1[tid];
  if (tid < DD) s_bo2[tid] = b_out2[tid];

  for (int i = tid; i < 16 * DD; i += 128) {
    int r = i >> 6, c = i & 63;
    float zv = z[(size_t)(blk * 16 + r) * DD + c];
    float vv = v_comb[(size_t)(blk * 16 + r) * DD + c];
    s_z0[i] = zv; s_vc[i] = vv;
    s_x[r * XS + c] = f32_to_bf16(zv);
  }
  for (int i = tid; i < 16 * HH; i += 128) {
    int r = i >> 7, c = i & 127;
    s_h[r * XS + c] = f32_to_bf16(h0[c]);
  }

  int ug0 = w * 2, ug1 = w * 2 + 1;
  float hold0[8], hold1[8], traj[8];
#pragma unroll
  for (int r = 0; r < 8; ++r) {
    hold0[r] = h0[ug0 * 16 + r + 8 * half];
    hold1[r] = h0[ug1 * 16 + r + 8 * half];
    traj[r]  = 0.f;
  }

  const v16bf* Aih = (const v16bf*)wf_ih;
  const v16bf* Ahh = (const v16bf*)wf_hh;
  const v16bf* Ao1 = (const v16bf*)wf_o1;
  const v16bf* Ao2 = (const v16bf*)wf_o2;

  __syncthreads();

  for (int step = 0; step < TT; ++step) {
    for (int i = tid; i < 16 * DD; i += 128) {
      int r = i >> 6, c = i & 63;
      int aid = actions[(blk * 16 + r) * TT + step];
      s_x[r * XS + DD + c] = f32_to_bf16(act_emb[(size_t)aid * DD + c]);
    }
    __syncthreads();

    v16bf bx[4], bh[4];
#pragma unroll
    for (int f = 0; f < 4; ++f) {
      bx[f] = build_bfrag(s_x, nl, half, f);
      bh[f] = build_bfrag(s_h, nl, half, f);
    }
    __syncthreads();  // all s_h reads done before h2 writes below

    float rg0[8], rg1[8], ug_g0[8], ug_g1[8], ng0[8], ng1[8];
    gate_slice(Aih, Ahh, bx, bh, ug0,     lane, s_bih,      s_bhh,      ug0*16, half, rg0);
    gate_slice(Aih, Ahh, bx, bh, ug1,     lane, s_bih,      s_bhh,      ug1*16, half, rg1);
    gate_slice(Aih, Ahh, bx, bh, ug0 + 8, lane, s_bih + HH, s_bhh + HH, ug0*16, half, ug_g0);
    gate_slice(Aih, Ahh, bx, bh, ug1 + 8, lane, s_bih + HH, s_bhh + HH, ug1*16, half, ug_g1);
    {
      AccU ai, ah;
#pragma unroll
      for (int r = 0; r < 8; ++r) {
        int unit = ug0 * 16 + r + 8 * half;
        ai.f[r] = s_bih[2 * HH + unit];
        ah.f[r] = s_bhh[2 * HH + unit];
      }
#pragma unroll
      for (int f = 0; f < 4; ++f) {
        v16bf wa = Aih[((size_t)(ug0 + 16) * 4 + f) * 32 + lane];
        v16bf wh = Ahh[((size_t)(ug0 + 16) * 4 + f) * 32 + lane];
        ai.v = __builtin_amdgcn_wmma_f32_16x16x32_bf16(false, wa, false, bx[f], (short)0, ai.v, false, false);
        ah.v = __builtin_amdgcn_wmma_f32_16x16x32_bf16(false, wh, false, bh[f], (short)0, ah.v, false, false);
      }
#pragma unroll
      for (int r = 0; r < 8; ++r) {
        float rr = 1.f / (1.f + expf(-rg0[r]));
        ng0[r] = tanhf(ai.f[r] + rr * ah.f[r]);
      }
    }
    {
      AccU ai, ah;
#pragma unroll
      for (int r = 0; r < 8; ++r) {
        int unit = ug1 * 16 + r + 8 * half;
        ai.f[r] = s_bih[2 * HH + unit];
        ah.f[r] = s_bhh[2 * HH + unit];
      }
#pragma unroll
      for (int f = 0; f < 4; ++f) {
        v16bf wa = Aih[((size_t)(ug1 + 16) * 4 + f) * 32 + lane];
        v16bf wh = Ahh[((size_t)(ug1 + 16) * 4 + f) * 32 + lane];
        ai.v = __builtin_amdgcn_wmma_f32_16x16x32_bf16(false, wa, false, bx[f], (short)0, ai.v, false, false);
        ah.v = __builtin_amdgcn_wmma_f32_16x16x32_bf16(false, wh, false, bh[f], (short)0, ah.v, false, false);
      }
#pragma unroll
      for (int r = 0; r < 8; ++r) {
        float rr = 1.f / (1.f + expf(-rg1[r]));
        ng1[r] = tanhf(ai.f[r] + rr * ah.f[r]);
      }
    }
#pragma unroll
    for (int r = 0; r < 8; ++r) {
      float uu0 = 1.f / (1.f + expf(-ug_g0[r]));
      float h20 = (1.f - uu0) * ng0[r] + uu0 * hold0[r];
      hold0[r] = h20;
      s_h[nl * XS + ug0 * 16 + r + 8 * half] = f32_to_bf16(h20);
      float uu1 = 1.f / (1.f + expf(-ug_g1[r]));
      float h21 = (1.f - uu1) * ng1[r] + uu1 * hold1[r];
      hold1[r] = h21;
      s_h[nl * XS + ug1 * 16 + r + 8 * half] = f32_to_bf16(h21);
    }
    __syncthreads();

    v16bf bh2[4];
#pragma unroll
    for (int f = 0; f < 4; ++f) bh2[f] = build_bfrag(s_h, nl, half, f);
#pragma unroll
    for (int g = 0; g < 2; ++g) {
      int ug = w * 2 + g;
      AccU o1;
#pragma unroll
      for (int r = 0; r < 8; ++r) o1.f[r] = s_bo1[ug * 16 + r + 8 * half];
#pragma unroll
      for (int f = 0; f < 4; ++f) {
        v16bf wa = Ao1[((size_t)ug * 4 + f) * 32 + lane];
        o1.v = __builtin_amdgcn_wmma_f32_16x16x32_bf16(false, wa, false, bh2[f],
                                                       (short)0, o1.v, false, false);
      }
#pragma unroll
      for (int r = 0; r < 8; ++r) {
        float x = o1.f[r];
        float ge = 0.5f * x * (1.f + erff(x * 0.7071067811865475f));
        s_g[nl * XS + ug * 16 + r + 8 * half] = f32_to_bf16(ge);
      }
    }
    __syncthreads();

    v16bf bg[4];
#pragma unroll
    for (int f = 0; f < 4; ++f) bg[f] = build_bfrag(s_g, nl, half, f);
    AccU zn;
#pragma unroll
    for (int r = 0; r < 8; ++r) zn.f[r] = s_bo2[w * 16 + r + 8 * half];
#pragma unroll
    for (int f = 0; f < 4; ++f) {
      v16bf wa = Ao2[((size_t)w * 4 + f) * 32 + lane];
      zn.v = __builtin_amdgcn_wmma_f32_16x16x32_bf16(false, wa, false, bg[f],
                                                     (short)0, zn.v, false, false);
    }
#pragma unroll
    for (int r = 0; r < 8; ++r) {
      float v = zn.f[r];
      traj[r] += v;
      s_x[nl * XS + w * 16 + r + 8 * half] = f32_to_bf16(v);
    }
    __syncthreads();
  }

  float sq = 0.f;
#pragma unroll
  for (int r = 0; r < 8; ++r) {
    int unit = w * 16 + r + 8 * half;
    float tm   = traj[r] * (1.0f / TT);
    float diff = tm - (s_z0[nl * DD + unit] + s_vc[nl * DD + unit]);
    sq += diff * diff;
  }
  s_red[tid] = sq;
  __syncthreads();
  for (int s = 64; s > 0; s >>= 1) {
    if (tid < s) s_red[tid] += s_red[tid + s];
    __syncthreads();
  }
  if (tid == 0) cf_part[blk] = s_red[0];

  for (int i = tid; i < 16 * DD; i += 128) {
    int r = i >> 6, c = i & 63;
    float ze = bfv[c];
    for (int j = 0; j < DD; ++j) ze += s_z0[r * DD + j] * Wf[c * 2 * DD + j];
    for (int j = 0; j < DD; ++j) ze += s_vc[r * DD + j] * Wf[c * 2 * DD + DD + j];
    int b = blk * 16 + r;
    z_out[(size_t)b * DD + c] = (gap_norm[b] > TAU_F) ? ze : s_z0[r * DD + c];
  }
}

// ---- K5: deterministic final reduction of 256 block partials -> mean ----
__global__ void k5_reduce(const float* __restrict__ cf_part, float* __restrict__ out_loss) {
  __shared__ float s_red[256];
  int t = threadIdx.x;
  s_red[t] = cf_part[t];
  __syncthreads();
  for (int st = 128; st > 0; st >>= 1) {
    if (t < st) s_red[t] += s_red[t + st];
    __syncthreads();
  }
  if (t == 0) out_loss[0] = s_red[0] * (1.0f / ((float)B_ROWS * DD));
}

extern "C" void kernel_launch(void* const* d_in, const int* in_sizes, int n_in,
                              void* d_out, int out_size, void* d_ws, size_t ws_size,
                              hipStream_t stream) {
  (void)in_sizes; (void)n_in; (void)out_size; (void)ws_size;
  const float* z      = (const float*)d_in[0];
  /* d_in[1] = E is unused by the reference */
  const float* hot    = (const float*)d_in[2];
  const float* gap    = (const float*)d_in[3];
  const float* Wq     = (const float*)d_in[4];
  const float* bq     = (const float*)d_in[5];
  const float* Wf     = (const float*)d_in[6];
  const float* bfv    = (const float*)d_in[7];
  const float* mem    = (const float*)d_in[8];
  const float* Wk     = (const float*)d_in[9];
  const float* Wo     = (const float*)d_in[10];
  const float* ckeys  = (const float*)d_in[11];
  const float* cvals  = (const float*)d_in[12];
  const float* aemb   = (const float*)d_in[13];
  const float* W_ih   = (const float*)d_in[14];
  const float* W_hh   = (const float*)d_in[15];
  const float* b_ih   = (const float*)d_in[16];
  const float* b_hh   = (const float*)d_in[17];
  const float* h0     = (const float*)d_in[18];
  const float* W_out1 = (const float*)d_in[19];
  const float* b_out1 = (const float*)d_in[20];
  const float* W_out2 = (const float*)d_in[21];
  const float* b_out2 = (const float*)d_in[22];
  const int*   acts   = (const int*)d_in[23];

  char* ws = (char*)d_ws;
  size_t off = 0;
  auto alloc = [&](size_t bytes) -> size_t {
    size_t o = off;
    off = (off + bytes + 255) & ~(size_t)255;
    return o;
  };
  float*          M       = (float*)(ws + alloc(DD * DD * 4));
  float*          P       = (float*)(ws + alloc(DD * DD * 4));
  unsigned short* qn      = (unsigned short*)(ws + alloc((size_t)B_ROWS * DD * 2));
  unsigned short* knfrag  = (unsigned short*)(ws + alloc((size_t)CC * DD * 2));
  float*          v_mem   = (float*)(ws + alloc((size_t)B_ROWS * DD * 4));
  float*          v_comb  = (float*)(ws + alloc((size_t)B_ROWS * DD * 4));
  float*          cf_part = (float*)(ws + alloc(4096));
  unsigned short* wf_ih   = (unsigned short*)(ws + alloc((size_t)24 * 2048 * 2));
  unsigned short* wf_hh   = (unsigned short*)(ws + alloc((size_t)24 * 2048 * 2));
  unsigned short* wf_o1   = (unsigned short*)(ws + alloc((size_t)8 * 2048 * 2));
  unsigned short* wf_o2   = (unsigned short*)(ws + alloc((size_t)4 * 2048 * 2));
  float*          out     = (float*)d_out;

  k0a_P   <<<DD, DD, 0, stream>>>(Wk, mem, P);
  k0b_M   <<<DD, DD, 0, stream>>>(P, Wo, M);
  k1_query<<<B_ROWS / 4, 256, 0, stream>>>(z, hot, Wq, bq, M, qn, v_mem);
  k2_keys <<<CC / 16, 256, 0, stream>>>(ckeys, knfrag);
  k2w_frag<<<24, 128, 0, stream>>>(W_ih, wf_ih);
  k2w_frag<<<24, 128, 0, stream>>>(W_hh, wf_hh);
  k2w_frag<<<8, 128, 0, stream>>>(W_out1, wf_o1);
  k2w_frag<<<4, 128, 0, stream>>>(W_out2, wf_o2);
  k3_sims <<<B_ROWS / 16, 128, 0, stream>>>(qn, knfrag, cvals, v_mem, v_comb);
  k4_gru  <<<B_ROWS / 16, 128, 0, stream>>>(z, gap, v_comb, aemb, acts,
                                            wf_ih, wf_hh, wf_o1, wf_o2,
                                            b_ih, b_hh, h0, b_out1, b_out2,
                                            Wf, bfv, out, cf_part);
  k5_reduce<<<1, 256, 0, stream>>>(cf_part, out + (size_t)B_ROWS * DD);
}
